// EuclideanMetric_57543971832354
// MI455X (gfx1250) — compile-verified, matrix operand order checked by reading, not run on hardware
//
#include <hip/hip_runtime.h>
#include <hip/hip_bf16.h>
#include <math.h>
#include <stdint.h>

typedef __attribute__((ext_vector_type(16))) __bf16 v16bf;
typedef __attribute__((ext_vector_type(8)))  __bf16 v8bf;
typedef __attribute__((ext_vector_type(8)))  float  v8f;

#define N_ROWS 8192
#define KDIM   1024
#define TILE_K 32
#define NITER  (KDIM / TILE_K)        // 32
#define LDS_STRIDE 40                 // 32 bf16 + 16B pad; keeps 16B alignment

// fast-path block tile: 256(M) x 128(N), 8 waves of 64x64
#define BM 256
#define BN 128
// LDS byte layout per buffer: Ahi[256x40], Alo[256x40], Bhi[128x40], Blo[128x40]
#define OFF_ALO 20480u
#define OFF_BHI 40960u
#define OFF_BLO 51200u
#define BUF_BYTES 61440u
#define BUF_ELEMS 30720

// fallback block tile: 128 x 128
#define FB_TILE 128
#define FB_ARR (FB_TILE * LDS_STRIDE)

// ---------------- kernel 1: row norms into workspace ----------------
__global__ void __launch_bounds__(256)
row_norms_kernel(const float* __restrict__ x1, const float* __restrict__ x2,
                 float* __restrict__ ws) {
  const int row = blockIdx.x;  // 0 .. 2*N_ROWS-1
  const float* src = (row < N_ROWS) ? (x1 + (size_t)row * KDIM)
                                    : (x2 + (size_t)(row - N_ROWS) * KDIM);
  float s = 0.f;
  for (int i = threadIdx.x; i < KDIM; i += 256) {
    float v = src[i];
    s += v * v;
  }
  __shared__ float red[256];
  red[threadIdx.x] = s;
  __syncthreads();
  for (int off = 128; off > 0; off >>= 1) {
    if (threadIdx.x < off) red[threadIdx.x] += red[threadIdx.x + off];
    __syncthreads();
  }
  if (threadIdx.x == 0) ws[row] = red[0];
}

// ---------------- kernel 2: f32 -> (bf16 hi, bf16 lo) split ----------------
__global__ void __launch_bounds__(256)
split_bf16_kernel(const float* __restrict__ x, __bf16* __restrict__ hi,
                  __bf16* __restrict__ lo) {
  const size_t base = ((size_t)blockIdx.x * 256 + threadIdx.x) * 8;
  const float4 a = *(const float4*)(x + base);
  const float4 b = *(const float4*)(x + base + 4);
  const float v[8] = {a.x, a.y, a.z, a.w, b.x, b.y, b.z, b.w};
  v8bf vh, vl;
#pragma unroll
  for (int j = 0; j < 8; ++j) {
    __bf16 h = (__bf16)v[j];
    vh[j] = h;
    vl[j] = (__bf16)(v[j] - (float)h);
  }
  *(v8bf*)(hi + base) = vh;
  *(v8bf*)(lo + base) = vl;
}

// Async 16B global -> LDS copy (CDNA5, ASYNCcnt-tracked). GVS addressing:
// mem = SADDR(64b) + VADDR(32b offset); LDS dst byte address from VGPR.
__device__ __forceinline__ void async_copy16(uint32_t lds_byte, const __bf16* base,
                                             uint32_t goff_bytes) {
  asm volatile("global_load_async_to_lds_b128 %0, %1, %2"
               :: "v"(lds_byte), "v"(goff_bytes), "s"(base)
               : "memory");
}

// Load one 16x32 bf16 fragment (K-major rows) from LDS in the CDNA5 WMMA
// per-lane layout: lane L (row = L&15, half = L>>4) holds
// K = 8*half..8*half+7 and K = 16+8*half..16+8*half+7.
__device__ __forceinline__ v16bf load_frag(const __bf16* tile, int row, int half) {
  const __bf16* p = tile + row * LDS_STRIDE + 8 * half;
  v8bf c0 = *(const v8bf*)p;
  v8bf c1 = *(const v8bf*)(p + 16);
  v16bf f;
#pragma unroll
  for (int i = 0; i < 8; ++i) { f[i] = c0[i]; f[i + 8] = c1[i]; }
  return f;
}

// ---------------- kernel 3 (fast path): async-staged split-bf16 WMMA GEMM ----------------
// 256x128 block tile, 8 waves each computing 64x64 (4x4 fragments).
// Per wave per K-step: 32 ds_load_b128 vs 48 v_wmma (ratio 0.67).
__global__ void __launch_bounds__(256)
cdist_wmma_async_kernel(const __bf16* __restrict__ x1hi, const __bf16* __restrict__ x1lo,
                        const __bf16* __restrict__ x2hi, const __bf16* __restrict__ x2lo,
                        const float* __restrict__ ws, float* __restrict__ out) {
  __shared__ __attribute__((aligned(16))) __bf16 sm[2 * BUF_ELEMS];  // 120 KB

  const int tid  = threadIdx.x;
  const int lane = tid & 31;
  const int wid  = tid >> 5;
  const int wm   = wid & 3;    // 4 wave-rows  -> 64 output rows each
  const int wn   = wid >> 2;   // 2 wave-cols  -> 64 output cols each
  const int r    = lane & 15;
  const int half = lane >> 4;

  const int bm = blockIdx.x * BM;
  const int bn = blockIdx.y * BN;

  // Per-thread staging slot: 16B granule (row0, ce) replicated over arrays
  // and i-strides. A arrays: i=0..3 (+64 rows), B arrays: i=0..1.
  const uint32_t row0 = (uint32_t)tid >> 2;        // 0..63
  const uint32_t ce   = ((uint32_t)tid & 3) * 8;   // k-chunk elems 0,8,16,24
  const uint32_t lin  = (row0 * LDS_STRIDE + ce) * 2;  // bytes within array
  const uint32_t goffA0 = (((uint32_t)bm + row0) * KDIM + ce) * 2u;
  const uint32_t goffB0 = (((uint32_t)bn + row0) * KDIM + ce) * 2u;
  const uint32_t smbase = (uint32_t)(uintptr_t)sm;

#define STAGE(bufsel, kbytes)                                                   \
  {                                                                             \
    const uint32_t lb = smbase + (uint32_t)(bufsel) * BUF_BYTES + lin;          \
    _Pragma("unroll")                                                           \
    for (uint32_t i = 0; i < 4; ++i) {                                          \
      async_copy16(lb + i * 5120u,           x1hi, goffA0 + (kbytes) + i * 131072u); \
      async_copy16(lb + OFF_ALO + i * 5120u, x1lo, goffA0 + (kbytes) + i * 131072u); \
    }                                                                           \
    _Pragma("unroll")                                                           \
    for (uint32_t i = 0; i < 2; ++i) {                                          \
      async_copy16(lb + OFF_BHI + i * 5120u, x2hi, goffB0 + (kbytes) + i * 131072u); \
      async_copy16(lb + OFF_BLO + i * 5120u, x2lo, goffB0 + (kbytes) + i * 131072u); \
    }                                                                           \
  }

  v8f acc[4][4] = {};

  STAGE(0, 0u);  // prologue: batch 0 -> buffer 0

  for (int it = 0; it < NITER; ++it) {
    const int buf = it & 1;
    if (it + 1 < NITER) {
      // Buffer buf^1 was released by the barrier ending iteration it-1.
      STAGE(buf ^ 1, (uint32_t)(it + 1) * TILE_K * 2u);
      // 12 newest async ops = next batch; <=12 outstanding => batch landed.
      asm volatile("s_wait_asynccnt 0xc" ::: "memory");
    } else {
      asm volatile("s_wait_asynccnt 0x0" ::: "memory");
    }
    __syncthreads();  // publish this batch's LDS to all waves

    const __bf16* tAhi = sm + buf * BUF_ELEMS;
    const __bf16* tAlo = tAhi + (OFF_ALO / 2);
    const __bf16* tBhi = tAhi + (OFF_BHI / 2);
    const __bf16* tBlo = tAhi + (OFF_BLO / 2);

    // A fragments resident; B fragments streamed (each loaded once, used 4x).
    v16bf aHi[4], aLo[4];
#pragma unroll
    for (int fm = 0; fm < 4; ++fm) {
      const int arow = wm * 64 + fm * 16 + r;
      aHi[fm] = load_frag(tAhi, arow, half);
      aLo[fm] = load_frag(tAlo, arow, half);
    }
#pragma unroll
    for (int fn = 0; fn < 4; ++fn) {
      const int brow = wn * 64 + fn * 16 + r;   // x2 row == output column
      const v16bf bH = load_frag(tBhi, brow, half);
      const v16bf bL = load_frag(tBlo, brow, half);
      // cross ~= hi*hi + hi*lo + lo*hi  (lo*lo dropped; below f32 ulp)
#pragma unroll
      for (int fm = 0; fm < 4; ++fm) {
        acc[fm][fn] = __builtin_amdgcn_wmma_f32_16x16x32_bf16(
            false, aHi[fm], false, bH, (short)0, acc[fm][fn], false, false);
        acc[fm][fn] = __builtin_amdgcn_wmma_f32_16x16x32_bf16(
            false, aHi[fm], false, bL, (short)0, acc[fm][fn], false, false);
        acc[fm][fn] = __builtin_amdgcn_wmma_f32_16x16x32_bf16(
            false, aLo[fm], false, bH, (short)0, acc[fm][fn], false, false);
      }
    }
    __syncthreads();  // release buffer buf for batch it+2
  }
#undef STAGE

  // Fused epilogue: d2 = sq1[m] + sq2[n] - 2*cross, clamp, -log1p.
  // C/D layout: VGPR v, lane L -> row = v + 8*(L>>4), col = L&15.
  const float* sq1 = ws;
  const float* sq2 = ws + N_ROWS;
#pragma unroll
  for (int fm = 0; fm < 4; ++fm) {
#pragma unroll
    for (int fn = 0; fn < 4; ++fn) {
      const int col = bn + wn * 64 + fn * 16 + r;
      const float s2 = sq2[col];
#pragma unroll
      for (int v = 0; v < 8; ++v) {
        const int row = bm + wm * 64 + fm * 16 + v + 8 * half;
        float d2 = sq1[row] + s2 - 2.0f * acc[fm][fn][v];
        d2 = fmaxf(d2, 0.0f);
        out[(size_t)row * N_ROWS + col] = -log1pf(d2);
      }
    }
  }
}

// ---------------- kernel 3b (fallback): in-kernel split, no extra workspace ----------------
__global__ void __launch_bounds__(256)
cdist_wmma_kernel(const float* __restrict__ x1, const float* __restrict__ x2,
                  const float* __restrict__ ws, float* __restrict__ out) {
  __shared__ __attribute__((aligned(16))) __bf16 sAhi[FB_ARR];
  __shared__ __attribute__((aligned(16))) __bf16 sAlo[FB_ARR];
  __shared__ __attribute__((aligned(16))) __bf16 sBhi[FB_ARR];
  __shared__ __attribute__((aligned(16))) __bf16 sBlo[FB_ARR];

  const int tid  = threadIdx.x;
  const int lane = tid & 31;
  const int wid  = tid >> 5;
  const int wm   = wid & 3;
  const int wn   = wid >> 2;
  const int r    = lane & 15;
  const int half = lane >> 4;

  const int bm = blockIdx.x * FB_TILE;
  const int bn = blockIdx.y * FB_TILE;

  v8f acc[2][4] = {};

  for (int kb = 0; kb < KDIM; kb += TILE_K) {
    __syncthreads();
#pragma unroll
    for (int i = 0; i < 4; ++i) {
      const int idx = tid + i * 256;
      const int row = idx >> 3;
      const int c4  = (idx & 7) * 4;
      const float4 a = *(const float4*)(x1 + (size_t)(bm + row) * KDIM + kb + c4);
      const float4 b = *(const float4*)(x2 + (size_t)(bn + row) * KDIM + kb + c4);
      const float av[4] = {a.x, a.y, a.z, a.w};
      const float bv[4] = {b.x, b.y, b.z, b.w};
#pragma unroll
      for (int j = 0; j < 4; ++j) {
        __bf16 h = (__bf16)av[j];
        sAhi[row * LDS_STRIDE + c4 + j] = h;
        sAlo[row * LDS_STRIDE + c4 + j] = (__bf16)(av[j] - (float)h);
        __bf16 g = (__bf16)bv[j];
        sBhi[row * LDS_STRIDE + c4 + j] = g;
        sBlo[row * LDS_STRIDE + c4 + j] = (__bf16)(bv[j] - (float)g);
      }
    }
    __syncthreads();

    v16bf aHi[2], aLo[2], bHi[4], bLo[4];
#pragma unroll
    for (int fm = 0; fm < 2; ++fm) {
      const int arow = wm * 32 + fm * 16 + r;
      aHi[fm] = load_frag(sAhi, arow, half);
      aLo[fm] = load_frag(sAlo, arow, half);
    }
#pragma unroll
    for (int fn = 0; fn < 4; ++fn) {
      const int brow = wn * 64 + fn * 16 + r;
      bHi[fn] = load_frag(sBhi, brow, half);
      bLo[fn] = load_frag(sBlo, brow, half);
    }
#pragma unroll
    for (int fm = 0; fm < 2; ++fm) {
#pragma unroll
      for (int fn = 0; fn < 4; ++fn) {
        acc[fm][fn] = __builtin_amdgcn_wmma_f32_16x16x32_bf16(
            false, aHi[fm], false, bHi[fn], (short)0, acc[fm][fn], false, false);
        acc[fm][fn] = __builtin_amdgcn_wmma_f32_16x16x32_bf16(
            false, aHi[fm], false, bLo[fn], (short)0, acc[fm][fn], false, false);
        acc[fm][fn] = __builtin_amdgcn_wmma_f32_16x16x32_bf16(
            false, aLo[fm], false, bHi[fn], (short)0, acc[fm][fn], false, false);
      }
    }
  }

  const float* sq1 = ws;
  const float* sq2 = ws + N_ROWS;
#pragma unroll
  for (int fm = 0; fm < 2; ++fm) {
#pragma unroll
    for (int fn = 0; fn < 4; ++fn) {
      const int col = bn + wn * 64 + fn * 16 + r;
      const float s2 = sq2[col];
#pragma unroll
      for (int v = 0; v < 8; ++v) {
        const int row = bm + wm * 32 + fm * 16 + v + 8 * half;
        float d2 = sq1[row] + s2 - 2.0f * acc[fm][fn][v];
        d2 = fmaxf(d2, 0.0f);
        out[(size_t)row * N_ROWS + col] = -log1pf(d2);
      }
    }
  }
}

extern "C" void kernel_launch(void* const* d_in, const int* in_sizes, int n_in,
                              void* d_out, int out_size, void* d_ws, size_t ws_size,
                              hipStream_t stream) {
  (void)in_sizes; (void)n_in; (void)out_size;
  const float* x1 = (const float*)d_in[0];
  const float* x2 = (const float*)d_in[1];
  float* out = (float*)d_out;
  float* ws  = (float*)d_ws;   // [0..8191]=sq1, [8192..16383]=sq2

  row_norms_kernel<<<2 * N_ROWS, 256, 0, stream>>>(x1, x2, ws);

  const size_t NORM_BYTES  = (size_t)2 * N_ROWS * sizeof(float);     // 64 KB
  const size_t MAT_ELEMS   = (size_t)N_ROWS * KDIM;
  const size_t SPLIT_BYTES = 4 * MAT_ELEMS * sizeof(__bf16);         // 64 MB

  if (ws_size >= NORM_BYTES + SPLIT_BYTES) {
    __bf16* x1hi = (__bf16*)((char*)d_ws + NORM_BYTES);
    __bf16* x1lo = x1hi + MAT_ELEMS;
    __bf16* x2hi = x1lo + MAT_ELEMS;
    __bf16* x2lo = x2hi + MAT_ELEMS;
    const int sgrid = (int)(MAT_ELEMS / (256 * 8));                  // 4096
    split_bf16_kernel<<<sgrid, 256, 0, stream>>>(x1, x1hi, x1lo);
    split_bf16_kernel<<<sgrid, 256, 0, stream>>>(x2, x2hi, x2lo);
    dim3 grid(N_ROWS / BM, N_ROWS / BN);                             // 32 x 64
    cdist_wmma_async_kernel<<<grid, 256, 0, stream>>>(x1hi, x1lo, x2hi, x2lo, ws, out);
  } else {
    dim3 grid(N_ROWS / FB_TILE, N_ROWS / FB_TILE);                   // 64 x 64
    cdist_wmma_kernel<<<grid, 256, 0, stream>>>(x1, x2, ws, out);
  }
}
